// Block_84585085927723
// MI455X (gfx1250) — matrix operand order repaired and checked by reading, not compiled, not run on hardware
//
#include <hip/hip_runtime.h>
#include <math.h>

// ---------------------------------------------------------------------------
// MI455X (gfx1250) implementation of the point-transformer block.
// wave32, WMMA f32_16x16x32_f16 for all GEMMs, f16 activations in workspace,
// async global->LDS staging (GLOBAL_LOAD_ASYNC_TO_LDS_B128) when available.
// ---------------------------------------------------------------------------

typedef __attribute__((ext_vector_type(16))) _Float16 v16h;
typedef __attribute__((ext_vector_type(8)))  _Float16 v8h;
typedef __attribute__((ext_vector_type(8)))  float    v8f;
typedef __attribute__((ext_vector_type(4)))  int      v4i;

#define NTOK   32768      // B*N = 4*8192
#define CDIM   256
#define NPTS   8192

#if __has_builtin(__builtin_amdgcn_global_load_async_to_lds_b128)
#define HAVE_ASYNC_LDS 1
#else
#define HAVE_ASYNC_LDS 0
#endif

// builtin prototype (from hipcc diagnostic): (AS1 v4i*, AS3 v4i*, imm, imm)
#define AS1V4(p) ((__attribute__((address_space(1))) v4i*)(p))
#define AS3V4(p) ((__attribute__((address_space(3))) v4i*)(p))

__device__ __forceinline__ void wait_async0() {
#if HAVE_ASYNC_LDS
#if __has_builtin(__builtin_amdgcn_s_wait_asynccnt)
  __builtin_amdgcn_s_wait_asynccnt(0);
#else
  asm volatile("s_wait_asynccnt 0x0" ::: "memory");
#endif
#endif
}

// ------------------------------ utility -----------------------------------

__global__ __launch_bounds__(256) void f32_to_f16_kernel(
    const float* __restrict__ in, _Float16* __restrict__ out, int n)
{
  int i = blockIdx.x * 256 + threadIdx.x;
  if (i < n) out[i] = (_Float16)in[i];
}

// ----------------------------- layernorm -----------------------------------
// one wave per row of 256; lane owns 8 channels; wave32 shuffle reduction.

__global__ __launch_bounds__(256) void ln_kernel(
    const float* __restrict__ x, const float* __restrict__ gam,
    const float* __restrict__ bet, _Float16* __restrict__ out)
{
  const int lane = threadIdx.x & 31;
  const int wid  = threadIdx.x >> 5;
  const size_t row = (size_t)blockIdx.x * 8 + wid;
  const float* xr = x + row * CDIM;

  float v[8];
  float4 a0 = *(const float4*)(xr + lane * 8);
  float4 a1 = *(const float4*)(xr + lane * 8 + 4);
  v[0]=a0.x; v[1]=a0.y; v[2]=a0.z; v[3]=a0.w;
  v[4]=a1.x; v[5]=a1.y; v[6]=a1.z; v[7]=a1.w;

  float s = 0.f;
  #pragma unroll
  for (int j = 0; j < 8; ++j) s += v[j];
  #pragma unroll
  for (int o = 16; o > 0; o >>= 1) s += __shfl_xor(s, o);
  const float mean = s * (1.f / 256.f);

  float q = 0.f;
  #pragma unroll
  for (int j = 0; j < 8; ++j) { float d = v[j] - mean; q = fmaf(d, d, q); }
  #pragma unroll
  for (int o = 16; o > 0; o >>= 1) q += __shfl_xor(q, o);
  const float rstd = rsqrtf(q * (1.f / 256.f) + 1e-5f);

  v8h o8;
  #pragma unroll
  for (int j = 0; j < 8; ++j) {
    int c = lane * 8 + j;
    o8[j] = (_Float16)((v[j] - mean) * rstd * gam[c] + bet[c]);
  }
  *(v8h*)(out + row * CDIM + lane * 8) = o8;
}

// ----------------------------- attention ------------------------------------
// one wave per token. lane owns 8 channels (lane*8..). Heads: lanes[8h..8h+7].
// Produces: xv = sum_k attn*v   (f16, 256/token)
//           S  = per-head attn-combined relu rows (f16, 4*256/token)
// pe is folded algebraically: sum_k attn*pe = S_h @ w_d2 + b_d2 (done in GEMM2)

__global__ __launch_bounds__(256) void attn_kernel(
    const float* __restrict__ xyz, const int* __restrict__ nns,
    const _Float16* __restrict__ qkv,
    const float* __restrict__ wd1, const float* __restrict__ bd1,
    _Float16* __restrict__ xv, _Float16* __restrict__ S)
{
  const int lane  = threadIdx.x & 31;
  const int wid   = threadIdx.x >> 5;
  const int point = blockIdx.x * 8 + wid;          // 0..32767
  const int b     = point >> 13;
  const size_t tok = (size_t)point;

  // q channels for this lane
  float qv[8];
  {
    v8h q8 = *(const v8h*)(qkv + tok * 768 + lane * 8);
    #pragma unroll
    for (int j = 0; j < 8; ++j) qv[j] = (float)q8[j];
  }

  // neighbor index + relative coords (lanes 0..15 own one neighbor each)
  int   idx = 0; float rxl = 0.f, ryl = 0.f;
  if (lane < 16) {
    idx = nns[tok * 32 + lane];
    float x0 = xyz[tok * 2 + 0], y0 = xyz[tok * 2 + 1];
    size_t nb = (size_t)(b * NPTS + idx);
    rxl = x0 - xyz[nb * 2 + 0];
    ryl = y0 - xyz[nb * 2 + 1];
  }
  int   idxA[16]; float rxA[16], ryA[16];
  #pragma unroll
  for (int k = 0; k < 16; ++k) {
    idxA[k] = __shfl(idx, k);
    rxA[k]  = __shfl(rxl, k);
    ryA[k]  = __shfl(ryl, k);
  }

  // attention scores: per-head dot over hd=64 (8 lanes x 8 ch, xor-reduce)
  float sc[16];
  #pragma unroll
  for (int k = 0; k < 16; ++k) {
    size_t nb = (size_t)(b * NPTS + idxA[k]);
    // prefetch the v row we will gather later (global_prefetch_b8)
    __builtin_prefetch(qkv + nb * 768 + 512 + lane * 8, 0, 1);
    v8h k8 = *(const v8h*)(qkv + nb * 768 + 256 + lane * 8);
    float p = 0.f;
    #pragma unroll
    for (int j = 0; j < 8; ++j) p = fmaf(qv[j], (float)k8[j], p);
    p += __shfl_xor(p, 1);
    p += __shfl_xor(p, 2);
    p += __shfl_xor(p, 4);
    sc[k] = p * 0.125f;   // 1/sqrt(64)
  }
  float m = sc[0];
  #pragma unroll
  for (int k = 1; k < 16; ++k) m = fmaxf(m, sc[k]);
  float attn[16]; float sum = 0.f;
  #pragma unroll
  for (int k = 0; k < 16; ++k) { attn[k] = __expf(sc[k] - m); sum += attn[k]; }
  const float inv = 1.f / sum;
  #pragma unroll
  for (int k = 0; k < 16; ++k) attn[k] *= inv;

  // xv = sum_k attn * v
  float acc[8] = {0.f,0.f,0.f,0.f,0.f,0.f,0.f,0.f};
  #pragma unroll
  for (int k = 0; k < 16; ++k) {
    size_t nb = (size_t)(b * NPTS + idxA[k]);
    v8h v8 = *(const v8h*)(qkv + nb * 768 + 512 + lane * 8);
    #pragma unroll
    for (int j = 0; j < 8; ++j) acc[j] = fmaf(attn[k], (float)v8[j], acc[j]);
  }
  {
    v8h o8;
    #pragma unroll
    for (int j = 0; j < 8; ++j) o8[j] = (_Float16)acc[j];
    *(v8h*)(xv + tok * CDIM + lane * 8) = o8;
  }

  // S[h, :] = sum_k attn[h,k] * relu(rel_k @ w_d1 + b_d1)
  const int head = lane >> 3;
  const int jb   = (lane & 7) * 32;
  for (int j = 0; j < 32; ++j) {
    int c = jb + j;
    float w0 = wd1[c], w1 = wd1[256 + c], bb = bd1[c];
    float s = 0.f;
    #pragma unroll
    for (int k = 0; k < 16; ++k) {
      float r = fmaf(rxA[k], w0, fmaf(ryA[k], w1, bb));
      r = fmaxf(r, 0.f);
      s = fmaf(attn[k], r, s);
    }
    S[tok * 1024 + head * 256 + c] = (_Float16)s;
  }
}

// ------------------------------ WMMA GEMM -----------------------------------
// C[M,N] = A[M,K](f16) @ B[K,N](f16), fused epilogues.
// Block: 256 threads = 8 waves (4x2), block tile 128x64, per-wave 32x32,
// K-chunk 32 -> one v_wmma_f32_16x16x32_f16 per 16x16 sub-tile per chunk.
// A tile is staged via GLOBAL_LOAD_ASYNC_TO_LDS_B128 (ASYNCcnt) when the
// toolchain exposes the builtin; B is transposed into LDS via registers.
// EPI: 0 = store f16
//      1 = d + bias + res16      -> f16
//      2 = d + bias + res32      -> f32
//      3 = gelu(d + bias)        -> f16
// aHeadStride: for the pe GEMM, A row slice depends on head = col/64.

template<int EPI>
__global__ __launch_bounds__(256) void gemm16_kernel(
    const _Float16* __restrict__ A, int lda,
    const _Float16* __restrict__ B, int ldb,
    int Kdim,
    const float* __restrict__ bias,
    const _Float16* __restrict__ res16,
    const float* __restrict__ res32,
    _Float16* __restrict__ out16, float* __restrict__ out32,
    int ldc, int aHeadStride)
{
  constexpr int BM = 128, BN = 64, KB = 32;
  __shared__ _Float16 sA[BM * KB];   // 8 KB, row-major [128][32]
  __shared__ _Float16 sBt[BN * KB];  // 4 KB, transposed [64][32] (K-major per col)

  const int tid  = threadIdx.x;
  const int lane = tid & 31;
  const int wave = tid >> 5;
  const int wm   = wave & 3;         // 4 waves along M
  const int wn   = wave >> 2;        // 2 waves along N
  const int hi   = lane >> 4;        // lane half (K-half selector)
  const int l15  = lane & 15;

  const int mTile = blockIdx.x * BM;
  const int nTile = blockIdx.y * BN;
  const int aOff  = (nTile / 64) * aHeadStride;

  // staging assignments (uniform across iterations)
  const int arow = tid >> 1, aseg = tid & 1;         // A: (row, 16-half segment)
  const int brow = tid >> 3, bseg = tid & 7;         // B: (K row, 8-col segment)

  const v8f vzero = {0.f,0.f,0.f,0.f,0.f,0.f,0.f,0.f};
  v8f acc[2][2];
  acc[0][0] = vzero; acc[0][1] = vzero; acc[1][0] = vzero; acc[1][1] = vzero;

  for (int k0 = 0; k0 < Kdim; k0 += KB) {
    // ---- prefetch next K-chunk source lines into caches
    if (k0 + KB < Kdim) {
      __builtin_prefetch(A + (size_t)(mTile + arow) * lda + aOff + k0 + KB + aseg * 16, 0, 1);
      __builtin_prefetch(B + (size_t)(k0 + KB + brow) * ldb + nTile + bseg * 8, 0, 1);
    }
    // ---- stage A tile (async global->LDS when available)
    {
      const _Float16* src = A + (size_t)(mTile + arow) * lda + aOff + k0 + aseg * 16;
      _Float16* dst = &sA[arow * KB + aseg * 16];
#if HAVE_ASYNC_LDS
      __builtin_amdgcn_global_load_async_to_lds_b128(AS1V4(src),     AS3V4(dst),     0, 0);
      __builtin_amdgcn_global_load_async_to_lds_b128(AS1V4(src + 8), AS3V4(dst + 8), 0, 0);
#else
      v8h a0 = *(const v8h*)(src);
      v8h a1 = *(const v8h*)(src + 8);
      *(v8h*)(dst)     = a0;
      *(v8h*)(dst + 8) = a1;
#endif
    }
    // ---- stage B tile transposed (half-granularity scatter via registers)
    {
      const _Float16* src = B + (size_t)(k0 + brow) * ldb + nTile + bseg * 8;
      v8h bv = *(const v8h*)src;
      #pragma unroll
      for (int j = 0; j < 8; ++j) sBt[(bseg * 8 + j) * KB + brow] = bv[j];
    }
    wait_async0();
    __syncthreads();

    // ---- build fragments per ISA wave32 layouts
    v16h afr[2], bfr[2];
    #pragma unroll
    for (int i = 0; i < 2; ++i) {
      // A 16x32: lane l -> row M=l15; halves 0..7 = K[hi*8..], 8..15 = K[16+hi*8..]
      int row = wm * 32 + i * 16 + l15;
      const _Float16* p = &sA[row * KB + hi * 8];
      v8h lo  = *(const v8h*)(p);
      v8h hi8 = *(const v8h*)(p + 16);
      afr[i] = __builtin_shufflevector(lo, hi8, 0,1,2,3,4,5,6,7,8,9,10,11,12,13,14,15);
    }
    #pragma unroll
    for (int i = 0; i < 2; ++i) {
      // B 32x16: lane l -> col N=l15; halves 0..7 = K[hi*8..], 8..15 = K[16+hi*8..]
      int col = wn * 32 + i * 16 + l15;
      const _Float16* p = &sBt[col * KB + hi * 8];
      v8h lo  = *(const v8h*)(p);
      v8h hi8 = *(const v8h*)(p + 16);
      bfr[i] = __builtin_shufflevector(lo, hi8, 0,1,2,3,4,5,6,7,8,9,10,11,12,13,14,15);
    }

    #pragma unroll
    for (int mi = 0; mi < 2; ++mi)
      #pragma unroll
      for (int ni = 0; ni < 2; ++ni)
        acc[mi][ni] = __builtin_amdgcn_wmma_f32_16x16x32_f16(
            false, afr[mi], false, bfr[ni], (short)0, acc[mi][ni], false, false);

    __syncthreads();
  }

  // ---- epilogue. D layout: VGPR e -> M = hi*8 + e, N = l15
  #pragma unroll
  for (int mi = 0; mi < 2; ++mi) {
    #pragma unroll
    for (int ni = 0; ni < 2; ++ni) {
      const int gcol = nTile + wn * 32 + ni * 16 + l15;
      #pragma unroll
      for (int e = 0; e < 8; ++e) {
        const int grow = mTile + wm * 32 + mi * 16 + hi * 8 + e;
        const size_t oi = (size_t)grow * ldc + gcol;
        float d = acc[mi][ni][e];
        if constexpr (EPI == 0) {
          out16[oi] = (_Float16)d;
        } else if constexpr (EPI == 1) {
          out16[oi] = (_Float16)(d + bias[gcol] + (float)res16[oi]);
        } else if constexpr (EPI == 2) {
          out32[oi] = d + bias[gcol] + res32[oi];
        } else {
          float t = d + bias[gcol];
          out16[oi] = (_Float16)(0.5f * t * (1.f + erff(t * 0.70710678118654752f)));
        }
      }
    }
  }
}

// ------------------------------- launcher -----------------------------------

extern "C" void kernel_launch(void* const* d_in, const int* in_sizes, int n_in,
                              void* d_out, int out_size, void* d_ws, size_t ws_size,
                              hipStream_t stream)
{
  (void)in_sizes; (void)n_in; (void)out_size; (void)ws_size;

  const float* xyz      = (const float*)d_in[0];
  const float* xy_embed = (const float*)d_in[1];
  const int*   nns      = (const int*)  d_in[2];
  const float* ln1_g    = (const float*)d_in[3];
  const float* ln1_b    = (const float*)d_in[4];
  const float* w_qkv    = (const float*)d_in[5];
  const float* w_d1     = (const float*)d_in[6];
  const float* b_d1     = (const float*)d_in[7];
  const float* w_d2     = (const float*)d_in[8];
  const float* b_d2     = (const float*)d_in[9];
  const float* w_proj   = (const float*)d_in[10];
  const float* b_proj   = (const float*)d_in[11];
  const float* ln2_g    = (const float*)d_in[12];
  const float* ln2_b    = (const float*)d_in[13];
  const float* w_m1     = (const float*)d_in[14];
  const float* b_m1     = (const float*)d_in[15];
  const float* w_m2     = (const float*)d_in[16];
  const float* b_m2     = (const float*)d_in[17];
  float* out = (float*)d_out;

  // ---- workspace layout (f16 region first, f32 tail); all offsets 512B-aligned
  _Float16* wqkv16  = (_Float16*)d_ws;
  _Float16* wd2_16  = wqkv16  + 256 * 768;
  _Float16* wproj16 = wd2_16  + 256 * 256;
  _Float16* wm1_16  = wproj16 + 256 * 256;
  _Float16* wm2_16  = wm1_16  + 256 * 1024;
  _Float16* h16     = wm2_16  + 1024 * 256;          // LN1 out, reused for LN2 out
  _Float16* qkv16   = h16     + (size_t)NTOK * 256;
  _Float16* S16     = qkv16   + (size_t)NTOK * 768;  // per-head combined relu rows
  _Float16* xv16    = S16     + (size_t)NTOK * 1024; // sum_k attn*v
  _Float16* xt16    = xv16    + (size_t)NTOK * 256;  // attn output incl. pe
  _Float16* g16     = xt16    + (size_t)NTOK * 256;  // gelu(h2@w_m1+b_m1)
  float*    x32     = (float*)(g16 + (size_t)NTOK * 1024); // post-proj residual

  // ---- weights -> f16
  auto cvt = [&](const float* src, _Float16* dst, int n) {
    f32_to_f16_kernel<<<(n + 255) / 256, 256, 0, stream>>>(src, dst, n);
  };
  cvt(w_qkv,  wqkv16,  256 * 768);
  cvt(w_d2,   wd2_16,  256 * 256);
  cvt(w_proj, wproj16, 256 * 256);
  cvt(w_m1,   wm1_16,  256 * 1024);
  cvt(w_m2,   wm2_16,  1024 * 256);

  // ---- LN1
  ln_kernel<<<NTOK / 8, 256, 0, stream>>>(xy_embed, ln1_g, ln1_b, h16);

  // ---- GEMM1: qkv = h @ w_qkv    (32768 x 256 x 768) -> f16
  {
    dim3 g(NTOK / 128, 768 / 64);
    gemm16_kernel<0><<<g, 256, 0, stream>>>(h16, 256, wqkv16, 768, 256,
        nullptr, nullptr, nullptr, qkv16, nullptr, 768, 0);
  }

  // ---- attention: xv + S
  attn_kernel<<<NTOK / 8, 256, 0, stream>>>(xyz, nns, qkv16, w_d1, b_d1, xv16, S16);

  // ---- GEMM2 (pe, head-sliced A): xt = S_h @ w_d2 + b_d2 + xv  -> f16
  {
    dim3 g(NTOK / 128, 256 / 64);
    gemm16_kernel<1><<<g, 256, 0, stream>>>(S16, 1024, wd2_16, 256, 256,
        b_d2, xv16, nullptr, xt16, nullptr, 256, /*aHeadStride=*/256);
  }

  // ---- GEMM3: x = xt @ w_proj + b_proj + xy_embed  -> f32
  {
    dim3 g(NTOK / 128, 256 / 64);
    gemm16_kernel<2><<<g, 256, 0, stream>>>(xt16, 256, wproj16, 256, 256,
        b_proj, nullptr, xy_embed, nullptr, x32, 256, 0);
  }

  // ---- LN2 (reuse h16)
  ln_kernel<<<NTOK / 8, 256, 0, stream>>>(x32, ln2_g, ln2_b, h16);

  // ---- GEMM4: g = gelu(h2 @ w_m1 + b_m1)  (32768 x 256 x 1024) -> f16
  {
    dim3 g(NTOK / 128, 1024 / 64);
    gemm16_kernel<3><<<g, 256, 0, stream>>>(h16, 256, wm1_16, 1024, 256,
        b_m1, nullptr, nullptr, g16, nullptr, 1024, 0);
  }

  // ---- GEMM5: out = g @ w_m2 + b_m2 + x  (32768 x 1024 x 256) -> f32
  {
    dim3 g(NTOK / 128, 256 / 64);
    gemm16_kernel<2><<<g, 256, 0, stream>>>(g16, 1024, wm2_16, 256, 1024,
        b_m2, nullptr, x32, nullptr, out, 256, 0);
  }
}
